// Encoder_2422361555422
// MI455X (gfx1250) — matrix operand order repaired and checked by reading, not compiled, not run on hardware
//
#include <hip/hip_runtime.h>
#include <hip/hip_bf16.h>

typedef unsigned int   u32;
typedef unsigned short u16;
typedef unsigned long long u64;
typedef __attribute__((ext_vector_type(16))) __bf16 bf16x16;
typedef __attribute__((ext_vector_type(8)))  float  f32x8;

union Frag { bf16x16 v; u32 d[8]; };

__device__ __forceinline__ u16 f2bf(float f) {
  union { float f; u32 u; } c; c.f = f;
  return (u16)((c.u + 0x7FFFu + ((c.u >> 16) & 1u)) >> 16);   // RNE
}
__device__ __forceinline__ float bf2f(u16 h) {
  union { u32 u; float f; } c; c.u = ((u32)h) << 16;
  return c.f;
}
__device__ __forceinline__ float gelu_f(float x) {
  return 0.5f * x * (1.0f + erff(x * 0.70710678118654752f));
}
__device__ __forceinline__ f32x8 wmma_bf16(bf16x16 a, bf16x16 b, f32x8 c) {
#if defined(__HIP_DEVICE_COMPILE__)
  return __builtin_amdgcn_wmma_f32_16x16x32_bf16(false, a, false, b, (short)0, c,
                                                 false, false);
#else
  return c;
#endif
}

// Async global->LDS copy of 16B per lane (CDNA5, tracked by ASYNCcnt).
__device__ __forceinline__ void async_b128(u32 lds_off, u64 gaddr) {
#if defined(__HIP_DEVICE_COMPILE__)
  asm volatile("global_load_async_to_lds_b128 %0, %1, off"
               :: "v"(lds_off), "v"(gaddr) : "memory");
#else
  (void)lds_off; (void)gaddr;
#endif
}
__device__ __forceinline__ void wait_async_le3() {
#if defined(__HIP_DEVICE_COMPILE__)
  asm volatile("s_wait_asynccnt 0x3" ::: "memory");
#endif
}
__device__ __forceinline__ void wait_async_0() {
#if defined(__HIP_DEVICE_COMPILE__)
  asm volatile("s_wait_asynccnt 0x0" ::: "memory");
#endif
}

// ---------------------------------------------------------------------------
// bf16 WMMA GEMM:  Out = epi(A[MxK] @ W[KxN] + bias [+ Res]), W pre-transposed
// (Wt is N x K).  Block: 256 threads = 8 waves, block tile 128x64, K-step 32,
// double-buffered GLOBAL_LOAD_ASYNC_TO_LDS_B128 staging. Each wave owns a
// 32x32 output tile (4 WMMA per K-step).
// epi: 0 = bias -> bf16, 1 = gelu(bias+acc) -> bf16, 2 = bias+acc+Res -> f32
// ---------------------------------------------------------------------------
__global__ __launch_bounds__(256) void gemm_bf16_k(
    const u16* __restrict__ A, const u16* __restrict__ Wt,
    const float* __restrict__ bias, const float* __restrict__ Res,
    u16* __restrict__ Outb, float* __restrict__ Outf,
    int M, int N, int K, int epi)
{
  __shared__ u16 sA[2][128][32];     // [buf][m][k]   8KB per buffer
  __shared__ u16 sB[2][64][32];      // [buf][n][k]   4KB per buffer
  const int tid  = threadIdx.x;
  const int m0   = blockIdx.x * 128;
  const int n0   = blockIdx.y * 64;
  const int wave = tid >> 5, lane = tid & 31;
  const int wm   = wave & 3, wn = wave >> 2;   // 4 x 2 waves of 32x32 tiles
  const int lx   = lane & 15, hf = lane >> 4;

  f32x8 acc00, acc01, acc10, acc11;
#pragma unroll
  for (int i = 0; i < 8; ++i) {
    acc00[i] = 0.0f; acc01[i] = 0.0f; acc10[i] = 0.0f; acc11[i] = 0.0f;
  }

  const int rowa = tid >> 2;            // 0..63
  const int segw = (tid & 3) * 8;       // bf16 element offset of 16B segment

  auto issue = [&](int k0, int buf) {
#pragma unroll
    for (int i = 0; i < 2; ++i) {       // A tile: 128 rows, 2 segs per thread
      int row  = rowa + i * 64;
      int grow = m0 + row; if (grow >= M) grow = M - 1;
      u64 ga = (u64)(size_t)(A + (size_t)grow * K + k0 + segw);
      u32 la = (u32)(size_t)&sA[buf][row][segw];
      async_b128(la, ga);
    }
    {                                   // B tile: 64 rows of Wt
      u64 ga = (u64)(size_t)(Wt + (size_t)(n0 + rowa) * K + k0 + segw);
      u32 la = (u32)(size_t)&sB[buf][rowa][segw];
      async_b128(la, ga);
    }
  };

  const int nkt = K >> 5;
  int buf = 0;
  issue(0, 0);
  for (int kt = 0; kt < nkt; ++kt) {
    if (kt + 1 < nkt) {
      issue((kt + 1) << 5, buf ^ 1);    // prefetch next tile into other buffer
      wait_async_le3();                 // current tile resident (in-order)
    } else {
      wait_async_0();
    }
    __syncthreads();

    Frag fa0, fa1, fb0, fb1;
    const int ml = wm * 32 + lx;
    const int nl = wn * 32 + lx;
#pragma unroll
    for (int r = 0; r < 4; ++r) {       // A frags (16x32, ISA layout)
      fa0.d[r]     = *(const u32*)&sA[buf][ml][hf * 8 + 2 * r];
      fa0.d[r + 4] = *(const u32*)&sA[buf][ml][16 + hf * 8 + 2 * r];
      fa1.d[r]     = *(const u32*)&sA[buf][ml + 16][hf * 8 + 2 * r];
      fa1.d[r + 4] = *(const u32*)&sA[buf][ml + 16][16 + hf * 8 + 2 * r];
    }
#pragma unroll
    for (int r = 0; r < 8; ++r) {       // B frags (32x16)
      fb0.d[r] = *(const u32*)&sB[buf][nl][hf * 8 + 2 * r + (r & 4) * 2];
      fb1.d[r] = *(const u32*)&sB[buf][nl + 16][hf * 8 + 2 * r + (r & 4) * 2];
    }
    acc00 = wmma_bf16(fa0.v, fb0.v, acc00);
    acc01 = wmma_bf16(fa0.v, fb1.v, acc01);
    acc10 = wmma_bf16(fa1.v, fb0.v, acc10);
    acc11 = wmma_bf16(fa1.v, fb1.v, acc11);
    __syncthreads();
    buf ^= 1;
  }

  const int col0  = n0 + wn * 32 + lx;
  const float bias0 = bias[col0];
  const float bias1 = bias[col0 + 16];
#pragma unroll
  for (int r = 0; r < 8; ++r) {
    int rbase = m0 + wm * 32 + r + 8 * hf;
#pragma unroll
    for (int a = 0; a < 2; ++a) {
      int row = rbase + a * 16;
      if (row >= M) continue;
      float v0 = (a ? acc10[r] : acc00[r]) + bias0;
      float v1 = (a ? acc11[r] : acc01[r]) + bias1;
      size_t i0 = (size_t)row * N + col0;
      size_t i1 = i0 + 16;
      if (epi == 2) {
        Outf[i0] = v0 + Res[i0];
        Outf[i1] = v1 + Res[i1];
      } else {
        if (epi == 1) { v0 = gelu_f(v0); v1 = gelu_f(v1); }
        Outb[i0] = f2bf(v0);
        Outb[i1] = f2bf(v1);
      }
    }
  }
}

// ---------------------------------------------------------------------------
// Fused flash attention, one wave per (16 q-rows, head, batch).
// Q,K,V bf16 in (B, n, 256) with head h at cols [h*64, h*64+64).
// ---------------------------------------------------------------------------
__global__ __launch_bounds__(32) void attn_k(
    const u16* __restrict__ Q, const u16* __restrict__ Kx,
    const u16* __restrict__ V, u16* __restrict__ O,
    int nq, int nk)
{
  __shared__ u16 sP[16][32];
  const int lane = threadIdx.x;
  const int lx = lane & 15, hf = lane >> 4;
  const int b = blockIdx.z, h = blockIdx.y, qt = blockIdx.x;
  const size_t qbase  = ((size_t)b * nq) * 256 + (size_t)h * 64;
  const size_t kvbase = ((size_t)b * nk) * 256 + (size_t)h * 64;

  Frag fq0, fq1;                                  // Q 16x64 = two A frags
  {
    int qrow = qt * 16 + lx; if (qrow >= nq) qrow = nq - 1;
    const u16* qp = Q + qbase + (size_t)qrow * 256;
#pragma unroll
    for (int r = 0; r < 4; ++r) {
      fq0.d[r]     = *(const u32*)(qp + hf * 8 + 2 * r);
      fq0.d[r + 4] = *(const u32*)(qp + 16 + hf * 8 + 2 * r);
      fq1.d[r]     = *(const u32*)(qp + 32 + hf * 8 + 2 * r);
      fq1.d[r + 4] = *(const u32*)(qp + 48 + hf * 8 + 2 * r);
    }
  }
  float rmax[8], rsum[8];
  f32x8 o0, o1, o2, o3;
#pragma unroll
  for (int i = 0; i < 8; ++i) {
    rmax[i] = -1e30f; rsum[i] = 0.0f;
    o0[i] = 0.0f; o1[i] = 0.0f; o2[i] = 0.0f; o3[i] = 0.0f;
  }

  const int nkt = (nk + 31) / 32;
  for (int kt = 0; kt < nkt; ++kt) {
    f32x8 s0, s1;
#pragma unroll
    for (int i = 0; i < 8; ++i) { s0[i] = 0.0f; s1[i] = 0.0f; }

#pragma unroll
    for (int sub = 0; sub < 2; ++sub) {           // S = Q @ K^T (16x32)
      int key = kt * 32 + sub * 16 + lx;
      if (key >= nk) key = nk - 1;
      const u16* kp = Kx + kvbase + (size_t)key * 256;
      Frag fk;
#pragma unroll
      for (int r = 0; r < 8; ++r) fk.d[r] = *(const u32*)(kp + hf * 8 + 2 * r + (r & 4) * 2);
      if (sub == 0) s0 = wmma_bf16(fq0.v, fk.v, s0);
      else          s1 = wmma_bf16(fq0.v, fk.v, s1);
#pragma unroll
      for (int r = 0; r < 8; ++r) fk.d[r] = *(const u32*)(kp + 32 + hf * 8 + 2 * r + (r & 4) * 2);
      if (sub == 0) s0 = wmma_bf16(fq1.v, fk.v, s0);
      else          s1 = wmma_bf16(fq1.v, fk.v, s1);
    }

#pragma unroll
    for (int r = 0; r < 8; ++r) {                 // online softmax per row
      float a0 = s0[r] * 0.125f;
      float a1 = s1[r] * 0.125f;
      if (kt * 32 + lx >= nk)      a0 = -1e30f;
      if (kt * 32 + 16 + lx >= nk) a1 = -1e30f;
      float rv = fmaxf(a0, a1);
#pragma unroll
      for (int m = 8; m >= 1; m >>= 1) rv = fmaxf(rv, __shfl_xor(rv, m, 32));
      float mnew = fmaxf(rmax[r], rv);
      float sc = __expf(rmax[r] - mnew);
      float p0 = __expf(a0 - mnew);
      float p1 = __expf(a1 - mnew);
      float ts = p0 + p1;
#pragma unroll
      for (int m = 8; m >= 1; m >>= 1) ts += __shfl_xor(ts, m, 32);
      rsum[r] = rsum[r] * sc + ts;
      rmax[r] = mnew;
      o0[r] *= sc; o1[r] *= sc; o2[r] *= sc; o3[r] *= sc;
      sP[r + 8 * hf][lx]      = f2bf(p0);         // C-layout -> LDS
      sP[r + 8 * hf][lx + 16] = f2bf(p1);
    }
    __syncthreads();

    Frag fp;                                      // LDS -> A-layout
#pragma unroll
    for (int r = 0; r < 4; ++r) {
      fp.d[r]     = *(const u32*)&sP[lx][hf * 8 + 2 * r];
      fp.d[r + 4] = *(const u32*)&sP[lx][16 + hf * 8 + 2 * r];
    }
#pragma unroll
    for (int f = 0; f < 4; ++f) {                 // O += P @ V
      Frag fv;
      const u16* vp = V + kvbase + f * 16 + lx;
#pragma unroll
      for (int r = 0; r < 8; ++r) {
        int ka  = kt * 32 + hf * 8 + 2 * r + (r & 4) * 2;
        int kb2 = ka + 1;
        if (ka  >= nk) ka  = nk - 1;
        if (kb2 >= nk) kb2 = nk - 1;
        u32 lo = vp[(size_t)ka * 256];
        u32 hi = vp[(size_t)kb2 * 256];
        fv.d[r] = lo | (hi << 16);
      }
      f32x8* op = (f == 0) ? &o0 : (f == 1) ? &o1 : (f == 2) ? &o2 : &o3;
      *op = wmma_bf16(fp.v, fv.v, *op);
    }
    __syncthreads();
  }

#pragma unroll
  for (int r = 0; r < 8; ++r) {
    int row = qt * 16 + r + 8 * hf;
    if (row >= nq) continue;
    float inv = 1.0f / rsum[r];
    u16* op = O + ((size_t)b * nq + row) * 256 + (size_t)h * 64 + lx;
    op[0]  = f2bf(o0[r] * inv);
    op[16] = f2bf(o1[r] * inv);
    op[32] = f2bf(o2[r] * inv);
    op[48] = f2bf(o3[r] * inv);
  }
}

// ---------------------------------------------------------------------------
// LayerNorm over C=256, one row per block; writes f32 and/or bf16.
// ---------------------------------------------------------------------------
__global__ __launch_bounds__(256) void ln_k(
    const float* __restrict__ X, const float* __restrict__ g,
    const float* __restrict__ bta, float* __restrict__ Yf, u16* __restrict__ Yb)
{
  __shared__ float red[256];
  const int row = blockIdx.x, tid = threadIdx.x;
  const size_t base = (size_t)row * 256;
  float x = X[base + tid];
  red[tid] = x;
  __syncthreads();
#pragma unroll
  for (int s = 128; s > 0; s >>= 1) {
    if (tid < s) red[tid] += red[tid + s];
    __syncthreads();
  }
  float mean = red[0] * (1.0f / 256.0f);
  __syncthreads();
  float d = x - mean;
  red[tid] = d * d;
  __syncthreads();
#pragma unroll
  for (int s = 128; s > 0; s >>= 1) {
    if (tid < s) red[tid] += red[tid + s];
    __syncthreads();
  }
  float var = red[0] * (1.0f / 256.0f);
  float y = d * rsqrtf(var + 1e-5f) * g[tid] + bta[tid];
  if (Yf) Yf[base + tid] = y;
  if (Yb) Yb[base + tid] = f2bf(y);
}

// ---------------------------------------------------------------------------
// Layout kernels: (B0,N,T,C) <-> (B0*T, N, C)
// ---------------------------------------------------------------------------
__global__ void trans_in_k(const float* __restrict__ X, float* __restrict__ xa,
                           u16* __restrict__ xb)
{
  size_t idx = (size_t)blockIdx.x * blockDim.x + threadIdx.x;
  const size_t total = (size_t)4 * 675 * 8 * 256;
  if (idx >= total) return;
  int c = (int)(idx & 255);
  size_t r = idx >> 8;
  int t = (int)(r & 7);
  size_t r2 = r >> 3;
  int n  = (int)(r2 % 675);
  int b0 = (int)(r2 / 675);
  size_t orow = ((size_t)(b0 * 8 + t)) * 675 + n;
  float v = X[idx];
  xa[orow * 256 + c] = v;
  xb[orow * 256 + c] = f2bf(v);
}

__global__ void trans_out_k(const float* __restrict__ xa, float* __restrict__ Y)
{
  size_t idx = (size_t)blockIdx.x * blockDim.x + threadIdx.x;
  const size_t total = (size_t)4 * 675 * 8 * 256;
  if (idx >= total) return;
  int c = (int)(idx & 255);
  size_t r = idx >> 8;
  int t = (int)(r & 7);
  size_t r2 = r >> 3;
  int n  = (int)(r2 % 675);
  int b0 = (int)(r2 / 675);
  size_t orow = ((size_t)(b0 * 8 + t)) * 675 + n;
  Y[idx] = xa[orow * 256 + c];
}

// Convert f32 weight (K x N) to transposed bf16 (N x K) for async-friendly GEMM.
__global__ void cvt_t_k(const float* __restrict__ X, u16* __restrict__ Y,
                        int K, int N)
{
  int i = blockIdx.x * blockDim.x + threadIdx.x;
  if (i >= K * N) return;
  int k = i / N, n = i - k * N;
  Y[(size_t)n * K + k] = f2bf(X[i]);
}

// ---------------------------------------------------------------------------
// Predictor helpers
// ---------------------------------------------------------------------------
__global__ __launch_bounds__(128) void glob_k(const u16* __restrict__ Hb,
                                              float* __restrict__ G)
{
  int b = blockIdx.x, j = threadIdx.x;
  float s = 0.0f;
  for (int n = 0; n < 675; ++n)
    s += bf2f(Hb[((size_t)b * 675 + n) * 256 + 128 + j]);
  G[b * 128 + j] = s * (1.0f / 675.0f);
}

__global__ void feat_k(const u16* __restrict__ Hb, const float* __restrict__ G,
                       u16* __restrict__ F)
{
  size_t idx = (size_t)blockIdx.x * blockDim.x + threadIdx.x;
  const size_t total = (size_t)21600 * 256;
  if (idx >= total) return;
  int c = (int)(idx & 255);
  size_t row = idx >> 8;
  int b = (int)(row / 675);
  F[idx] = (c < 128) ? Hb[idx] : f2bf(G[b * 128 + (c - 128)]);
}

__global__ void score_k(const u16* __restrict__ Z, const float* __restrict__ W3,
                        const float* __restrict__ b3, float* __restrict__ S, int M)
{
  int row = blockIdx.x * blockDim.x + threadIdx.x;
  if (row >= M) return;
  float l0 = b3[0], l1 = b3[1];
  for (int k = 0; k < 64; ++k) {
    float z = bf2f(Z[(size_t)row * 64 + k]);
    l0 += z * W3[k * 2];
    l1 += z * W3[k * 2 + 1];
  }
  float m = fmaxf(l0, l1);
  S[row] = l0 - (m + logf(__expf(l0 - m) + __expf(l1 - m)));
}

// Exact top-k by rank (matches jax top_k: descending score, stable by index).
__global__ __launch_bounds__(256) void topk_k(const float* __restrict__ S,
                                              int* __restrict__ keep)
{
  __shared__ float sc[675];
  int b = blockIdx.x;
  for (int n = threadIdx.x; n < 675; n += 256) sc[n] = S[b * 675 + n];
  __syncthreads();
  for (int n = threadIdx.x; n < 675; n += 256) {
    float v = sc[n];
    int rank = 0;
    for (int m = 0; m < 675; ++m) {
      float u = sc[m];
      rank += (u > v) || (u == v && m < n);
    }
    if (rank < 540) keep[b * 540 + rank] = n;
  }
}

__global__ void gather_k(const u16* __restrict__ Xb, const int* __restrict__ keep,
                         u16* __restrict__ Kr)
{
  size_t idx = (size_t)blockIdx.x * blockDim.x + threadIdx.x;
  const size_t total = (size_t)32 * 540 * 256;
  if (idx >= total) return;
  int c = (int)(idx & 255);
  size_t row = idx >> 8;
  int j = (int)(row % 540);
  int b = (int)(row / 540);
  int n = keep[b * 540 + j];
  Kr[idx] = Xb[((size_t)b * 675 + n) * 256 + c];
}

// ---------------------------------------------------------------------------
extern "C" void kernel_launch(void* const* d_in, const int* in_sizes, int n_in,
                              void* d_out, int out_size, void* d_ws, size_t ws_size,
                              hipStream_t stream) {
  (void)in_sizes; (void)n_in; (void)out_size; (void)ws_size;
  const float* x      = (const float*)d_in[0];
  const float* Wq     = (const float*)d_in[1];
  const float* bq     = (const float*)d_in[2];
  const float* Wk     = (const float*)d_in[3];
  const float* bk     = (const float*)d_in[4];
  const float* Wv     = (const float*)d_in[5];
  const float* bv     = (const float*)d_in[6];
  const float* Wo     = (const float*)d_in[7];
  const float* bo     = (const float*)d_in[8];
  const float* ln1_g  = (const float*)d_in[9];
  const float* ln1_b  = (const float*)d_in[10];
  const float* W1     = (const float*)d_in[11];
  const float* b1     = (const float*)d_in[12];
  const float* W2     = (const float*)d_in[13];
  const float* b2     = (const float*)d_in[14];
  const float* ln2_g  = (const float*)d_in[15];
  const float* ln2_b  = (const float*)d_in[16];
  const float* p_ln_g = (const float*)d_in[17];
  const float* p_ln_b = (const float*)d_in[18];
  const float* p_in_w = (const float*)d_in[19];
  const float* p_in_b = (const float*)d_in[20];
  const float* p_w1   = (const float*)d_in[21];
  const float* p_b1   = (const float*)d_in[22];
  const float* p_w2   = (const float*)d_in[23];
  const float* p_b2   = (const float*)d_in[24];
  const float* p_w3   = (const float*)d_in[25];
  const float* p_b3   = (const float*)d_in[26];

  const int Nn = 675, C = 256, FFd = 1024, KEEPN = 540;
  const size_t M1 = (size_t)32 * Nn;      // 21600
  const size_t M2 = (size_t)32 * KEEPN;   // 17280

  char* base = (char*)d_ws;
  size_t off = 0;
  auto carve = [&](size_t bytes) -> void* {
    off = (off + 255) & ~(size_t)255;
    void* p = base + off;
    off += bytes;
    return p;
  };
  float* xa   = (float*)carve(M1 * C * 4);
  u16*   xb   = (u16*)  carve(M1 * C * 2);
  u16*   qb   = (u16*)  carve(M1 * C * 2);
  u16*   kbuf = (u16*)  carve(M1 * C * 2);
  u16*   vbuf = (u16*)  carve(M1 * C * 2);
  u16*   obuf = (u16*)  carve(M1 * C * 2);
  float* tmpf = (float*)carve(M1 * C * 4);
  float* h1f  = (float*)carve(M1 * C * 4);
  u16*   h1b  = (u16*)  carve(M1 * C * 2);
  u16*   gbuf = (u16*)  carve(M1 * FFd * 2);
  u16*   krb  = (u16*)  carve(M2 * C * 2);
  u16*   wqb  = (u16*)  carve(65536 * 2);
  u16*   wkb  = (u16*)  carve(65536 * 2);
  u16*   wvb  = (u16*)  carve(65536 * 2);
  u16*   wob  = (u16*)  carve(65536 * 2);
  u16*   w1b  = (u16*)  carve(262144 * 2);
  u16*   w2b  = (u16*)  carve(262144 * 2);
  u16*   pinwb= (u16*)  carve(65536 * 2);
  u16*   pw1b = (u16*)  carve(32768 * 2);
  u16*   pw2b = (u16*)  carve(8192 * 2);
  float* glob = (float*)carve(32 * 128 * 4);
  float* score= (float*)carve(M1 * 4);
  int*   keep = (int*)  carve(32 * 540 * 4);
  // Predictor intermediates alias Q/K/V/O/h1b (dead before projections run).
  u16* lnpb = qb;  u16* hb = kbuf;  u16* featb = vbuf;
  u16* z1b  = obuf; u16* z2b = h1b;

  auto g1 = [](size_t n) { return (unsigned)((n + 255) / 256); };

  // Weights -> transposed bf16 (N x K), once per launch.
  cvt_t_k<<<g1(65536), 256, 0, stream>>>(Wq, wqb, 256, 256);
  cvt_t_k<<<g1(65536), 256, 0, stream>>>(Wk, wkb, 256, 256);
  cvt_t_k<<<g1(65536), 256, 0, stream>>>(Wv, wvb, 256, 256);
  cvt_t_k<<<g1(65536), 256, 0, stream>>>(Wo, wob, 256, 256);
  cvt_t_k<<<g1(262144), 256, 0, stream>>>(W1, w1b, 256, 1024);
  cvt_t_k<<<g1(262144), 256, 0, stream>>>(W2, w2b, 1024, 256);
  cvt_t_k<<<g1(65536), 256, 0, stream>>>(p_in_w, pinwb, 256, 256);
  cvt_t_k<<<g1(32768), 256, 0, stream>>>(p_w1, pw1b, 256, 128);
  cvt_t_k<<<g1(8192), 256, 0, stream>>>(p_w2, pw2b, 128, 64);

  trans_in_k<<<g1(M1 * C), 256, 0, stream>>>(x, xa, xb);

  const unsigned gM1 = (unsigned)((M1 + 127) / 128);
  const unsigned gM2 = (unsigned)((M2 + 127) / 128);
  const unsigned qtiles = (unsigned)((Nn + 15) / 16);

  for (int layer = 0; layer < 4; ++layer) {
    const u16* kvsrc = xb;
    unsigned gKV = gM1;
    int MkvI = (int)M1, nk = Nn;

    if (layer == 1) {
      // ---- token-pruning predictor ----
      ln_k<<<(unsigned)M1, 256, 0, stream>>>(xa, p_ln_g, p_ln_b, nullptr, lnpb);
      gemm_bf16_k<<<dim3(gM1, C / 64), 256, 0, stream>>>(
          lnpb, pinwb, p_in_b, nullptr, hb, nullptr, (int)M1, C, C, 1);
      glob_k<<<32, 128, 0, stream>>>(hb, glob);
      feat_k<<<g1(M1 * C), 256, 0, stream>>>(hb, glob, featb);
      gemm_bf16_k<<<dim3(gM1, 128 / 64), 256, 0, stream>>>(
          featb, pw1b, p_b1, nullptr, z1b, nullptr, (int)M1, 128, 256, 1);
      gemm_bf16_k<<<dim3(gM1, 64 / 64), 256, 0, stream>>>(
          z1b, pw2b, p_b2, nullptr, z2b, nullptr, (int)M1, 64, 128, 1);
      score_k<<<g1(M1), 256, 0, stream>>>(z2b, p_w3, p_b3, score, (int)M1);
      topk_k<<<32, 256, 0, stream>>>(score, keep);
      gather_k<<<g1(M2 * C), 256, 0, stream>>>(xb, keep, krb);
      kvsrc = krb; gKV = gM2; MkvI = (int)M2; nk = KEEPN;
    }

    // ---- QKV projections ----
    gemm_bf16_k<<<dim3(gM1, C / 64), 256, 0, stream>>>(
        xb, wqb, bq, nullptr, qb, nullptr, (int)M1, C, C, 0);
    gemm_bf16_k<<<dim3(gKV, C / 64), 256, 0, stream>>>(
        kvsrc, wkb, bk, nullptr, kbuf, nullptr, MkvI, C, C, 0);
    gemm_bf16_k<<<dim3(gKV, C / 64), 256, 0, stream>>>(
        kvsrc, wvb, bv, nullptr, vbuf, nullptr, MkvI, C, C, 0);

    // ---- fused flash attention ----
    attn_k<<<dim3(qtiles, 4, 32), 32, 0, stream>>>(qb, kbuf, vbuf, obuf, Nn, nk);

    // ---- output proj + residual, LN1 ----
    gemm_bf16_k<<<dim3(gM1, C / 64), 256, 0, stream>>>(
        obuf, wob, bo, xa, nullptr, tmpf, (int)M1, C, C, 2);
    ln_k<<<(unsigned)M1, 256, 0, stream>>>(tmpf, ln1_g, ln1_b, h1f, h1b);

    // ---- FFN + residual, LN2 ----
    gemm_bf16_k<<<dim3(gM1, FFd / 64), 256, 0, stream>>>(
        h1b, w1b, b1, nullptr, gbuf, nullptr, (int)M1, FFd, C, 1);
    gemm_bf16_k<<<dim3(gM1, C / 64), 256, 0, stream>>>(
        gbuf, w2b, b2, h1f, nullptr, tmpf, (int)M1, C, FFd, 2);
    ln_k<<<(unsigned)M1, 256, 0, stream>>>(tmpf, ln2_g, ln2_b, xa, xb);
  }

  trans_out_k<<<g1(M1 * C), 256, 0, stream>>>(xa, (float*)d_out);
}
